// BiDACPI_69011534512283
// MI455X (gfx1250) — matrix operand order repaired and checked by reading, not compiled, not run on hardware
//
#include <hip/hip_runtime.h>
#include <hip/hip_bf16.h>
#include <math.h>

// ---------------------------------------------------------------------------
// Types & helpers
// ---------------------------------------------------------------------------
typedef unsigned short u16;
typedef __attribute__((ext_vector_type(8)))  u16    v8u;
typedef __attribute__((ext_vector_type(16))) u16    v16u;
typedef __attribute__((ext_vector_type(16))) __bf16 v16bf;
typedef __attribute__((ext_vector_type(8)))  float  v8f;

union FragU { v16u u; v16bf b; };

#define ALPHA_ 0.2f
#define NEGF  -9.0e15f

__device__ __forceinline__ u16 f2bf(float f) {
  unsigned u = __builtin_bit_cast(unsigned, f);
  u = (u + 0x7FFFu + ((u >> 16) & 1u)) >> 16;   // round-to-nearest-even
  return (u16)u;
}
__device__ __forceinline__ float bf2f(u16 h) {
  unsigned u = ((unsigned)h) << 16;
  return __builtin_bit_cast(float, u);
}
__device__ __forceinline__ float leakyf(float x) { return x > 0.f ? x : ALPHA_ * x; }
__device__ __forceinline__ float eluf(float x)   { return x > 0.f ? x : (__expf(x) - 1.f); }

// ---------------------------------------------------------------------------
// Elementwise fp32 -> bf16 convert (row-major)
// ---------------------------------------------------------------------------
__global__ void k_f2bf(const float* __restrict__ in, u16* __restrict__ out, long long n) {
  long long i  = (long long)blockIdx.x * blockDim.x + threadIdx.x;
  long long st = (long long)gridDim.x * blockDim.x;
  for (; i < n; i += st) out[i] = f2bf(in[i]);
}

// ---------------------------------------------------------------------------
// Pack an fp32 KxN(=128) weight matrix into bf16 WMMA B-fragment order:
//   dst[(((kc*8 + nt)*2 + g)*16 + col)*16 + e] = W[(kc*32 + g*16 + e)*128 + nt*16 + col]
// so each lane's 16-element B fragment is one contiguous 32B run.
// Batched over blockIdx.y.
// ---------------------------------------------------------------------------
__global__ void k_pack_w(const float* __restrict__ src, u16* __restrict__ dst,
                         long long n, long long srcBatch, long long dstBatch) {
  const float* S = src + srcBatch * blockIdx.y;
  u16*         D = dst + dstBatch * blockIdx.y;
  long long i  = (long long)blockIdx.x * blockDim.x + threadIdx.x;
  long long st = (long long)gridDim.x * blockDim.x;
  for (; i < n; i += st) {
    int e = (int)(i & 15), cl = (int)((i >> 4) & 15);
    int g = (int)((i >> 8) & 1), nt = (int)((i >> 9) & 7);
    long long kc = i >> 12;
    long long k  = kc * 32 + g * 16 + e;
    D[i] = f2bf(S[k * 128 + nt * 16 + cl]);
  }
}

// ---------------------------------------------------------------------------
// Embedding gathers
// ---------------------------------------------------------------------------
__global__ void k_gather_bf(const int* __restrict__ idx, const float* __restrict__ emb,
                            u16* __restrict__ out, int C) {
  int row = blockIdx.x, c = threadIdx.x;
  out[(size_t)row * C + c] = f2bf(emb[(size_t)idx[row] * C + c]);
}
__global__ void k_gather_f32(const int* __restrict__ idx, const float* __restrict__ emb,
                             float* __restrict__ out, int C) {
  int row = blockIdx.x, c = threadIdx.x;
  out[(size_t)row * C + c] = emb[(size_t)idx[row] * C + c];
}

// ---------------------------------------------------------------------------
// WMMA GEMM:  Y(MxN=128) = X(MxK) @ W(KxN=128), bf16 in / f32 accumulate.
// 1 wave per block computes a 16x128 output tile.  No LDS:
//  - A fragments: two contiguous v8u global loads (contiguous k = 16h + s order, same as packed B)
//  - B fragments: one contiguous v16u global load from the packed weight
// act: 0=none, 1=leaky, 2=elu. Outputs: obf (row-major bf16), of32 (row-major
// fp32), opk (bf16 in packed-B layout for a downstream WMMA consumer); each
// may be null. Batched over blockIdx.y.
// ---------------------------------------------------------------------------
__global__ void k_gemm_wmma(const u16* __restrict__ Xg, const u16* __restrict__ Wp,
                            const float* __restrict__ bias,
                            u16* __restrict__ obf, float* __restrict__ of32,
                            u16* __restrict__ opk,
                            int M, int K,
                            long long xBatch, long long wBatch, long long oBatch,
                            int act) {
  const int lane = threadIdx.x;
  const int lo   = (lane < 16);
  const int colL = lane & 15;
  const u16* X = Xg + xBatch * blockIdx.y;
  const u16* W = Wp + wBatch * blockIdx.y;
  const long long ob = oBatch * blockIdx.y;
  const int row0 = blockIdx.x * 16;
  const int arow = row0 + colL;
  const size_t laneoff = (size_t)(((lo ? 0 : 16) + colL) * 16);

  v8f acc[8];
#pragma unroll
  for (int t = 0; t < 8; ++t)
#pragma unroll
    for (int r = 0; r < 8; ++r) acc[t][r] = 0.f;

  const int nChunks = K / 32;
  for (int kc = 0; kc < nChunks; ++kc) {
    FragU af;
    {
      const v8u va = *(const v8u*)&X[(size_t)arow * K + kc * 32 + (lo ? 0 : 16)];
      const v8u vb = *(const v8u*)&X[(size_t)arow * K + kc * 32 + (lo ? 8 : 24)];
      af.u = __builtin_shufflevector(va, vb, 0, 1, 2, 3, 4, 5, 6, 7,
                                             8, 9, 10, 11, 12, 13, 14, 15);
    }
#pragma unroll
    for (int nt = 0; nt < 8; ++nt) {
      FragU bf_;
      bf_.u = *(const v16u*)&W[((size_t)(kc * 8 + nt)) * 512 + laneoff];
      acc[nt] = __builtin_amdgcn_wmma_f32_16x16x32_bf16(
          false, af.b, false, bf_.b, (short)0, acc[nt], false, false);
    }
  }

  // epilogue: bias + activation + stores
#pragma unroll
  for (int nt = 0; nt < 8; ++nt) {
    int col = nt * 16 + colL;
    float bv = bias ? bias[col] : 0.f;
#pragma unroll
    for (int r = 0; r < 8; ++r) {
      int row = row0 + r + (lo ? 0 : 8);
      float v = acc[nt][r] + bv;
      if (act == 1)      v = leakyf(v);
      else if (act == 2) v = eluf(v);
      u16 hv = f2bf(v);
      if (obf)  obf[(size_t)ob + (size_t)row * 128 + col] = hv;
      if (of32) of32[(size_t)ob + (size_t)row * 128 + col] = v;
      if (opk) {
        // packed-B layout for the next WMMA stage (row becomes K)
        size_t pi = ((((size_t)(row >> 5) * 8 + nt) * 2 + ((row >> 4) & 1)) * 16 + colL) * 16
                    + (row & 15);
        opk[(size_t)ob + pi] = hv;
      }
    }
  }
}

// ---------------------------------------------------------------------------
// e1/e2 row projections:  e1[row]=Wh[row,:].a1 ; e2[row]=Wh[row,:].a2
// wave per row, shuffle reduction (reads row-major bf16 Wh).
// ---------------------------------------------------------------------------
__global__ void k_rowdot(const u16* __restrict__ X, const float* __restrict__ avec,
                         float* __restrict__ e1, float* __restrict__ e2,
                         int rowsPerA, int aPairStride, int nrows) {
  int wid = threadIdx.x >> 5, lane = threadIdx.x & 31;
  int row = blockIdx.x * (blockDim.x >> 5) + wid;
  if (row >= nrows) return;
  int h = row / rowsPerA;
  const float* a1 = avec + (size_t)h * aPairStride;
  const float* a2 = a1 + 128;
  float s1 = 0.f, s2 = 0.f;
#pragma unroll
  for (int j = 0; j < 4; ++j) {
    int f = lane + 32 * j;
    float x = bf2f(X[(size_t)row * 128 + f]);
    s1 += x * a1[f];
    s2 += x * a2[f];
  }
#pragma unroll
  for (int off = 16; off; off >>= 1) {
    s1 += __shfl_xor(s1, off, 32);
    s2 += __shfl_xor(s2, off, 32);
  }
  if (lane == 0) { e1[row] = s1; e2[row] = s2; }
}

// ---------------------------------------------------------------------------
// Fused GAT attention: per (h,b,16-row tile) masked softmax over 512 neighbors
// then att(16x32) @ Wh(32x128) accumulated via WMMA.  Wh is consumed in
// packed-B layout (one v16u global load per fragment); attention weights
// staged in LDS row-major and read as two ds_load_b128 per A fragment.
// ELU applied; out[(b*NA+n)*outStride + h*outColOff + f] (bf16).
// ---------------------------------------------------------------------------
__global__ void k_gat_attn(const u16* __restrict__ Whp, const float* __restrict__ e1g,
                           const float* __restrict__ e2g, const int* __restrict__ adj,
                           u16* __restrict__ out, int outStride, int outColOff,
                           int NA_) {
  __shared__ __align__(32) u16 As[16 * 32];
  __shared__ float e1s[16], rmax[16], rsum[16];
  const int lane = threadIdx.x;
  const int lo   = (lane < 16);
  const int colL = lane & 15;
  const int h = blockIdx.z, b = blockIdx.y, Bn = gridDim.y;
  const int n0 = blockIdx.x * 16;
  const size_t MAel    = (size_t)Bn * NA_ * 128;         // elements per h-batch
  const size_t whBase  = (size_t)h * MAel;               // packed Wh base for this head
  const size_t eBase   = ((size_t)(h * Bn + b)) * NA_;
  const size_t adjBase = (size_t)b * NA_ * NA_;
  const size_t laneoff = (size_t)(((lo ? 0 : 16) + colL) * 16);

  if (lane < 16) e1s[lane] = e1g[eBase + n0 + lane];
  __syncthreads();

  // softmax statistics: 2 lanes per row scan 256 neighbors each
  {
    const int r = colL;
    const int mstart = lo ? 0 : (NA_ / 2);
    const float e1v = e1s[r];
    const size_t arow = adjBase + (size_t)(n0 + r) * NA_;
    float mx = NEGF;
    for (int m = mstart; m < mstart + NA_ / 2; ++m) {
      float v = (adj[arow + m] > 0) ? leakyf(e1v + e2g[eBase + m]) : NEGF;
      mx = fmaxf(mx, v);
    }
    mx = fmaxf(mx, __shfl_xor(mx, 16, 32));
    float sm = 0.f;
    for (int m = mstart; m < mstart + NA_ / 2; ++m) {
      float v = (adj[arow + m] > 0) ? leakyf(e1v + e2g[eBase + m]) : NEGF;
      sm += __expf(v - mx);
    }
    sm += __shfl_xor(sm, 16, 32);
    if (lo) { rmax[r] = mx; rsum[r] = sm; }
  }
  __syncthreads();

  v8f acc[8];
#pragma unroll
  for (int t = 0; t < 8; ++t)
#pragma unroll
    for (int r = 0; r < 8; ++r) acc[t][r] = 0.f;

  const int nChunks = NA_ / 32;
  for (int kc = 0; kc < nChunks; ++kc) {
    // attention weights for this 32-neighbor chunk -> LDS (bf16, row-major)
    const int m = kc * 32 + lane;
    const float e2v = e2g[eBase + m];
#pragma unroll
    for (int r2 = 0; r2 < 16; ++r2) {
      float v = (adj[adjBase + (size_t)(n0 + r2) * NA_ + m] > 0)
                    ? leakyf(e1s[r2] + e2v) : NEGF;
      float w = __expf(v - rmax[r2]) / rsum[r2];
      As[r2 * 32 + lane] = f2bf(w);
    }
    __syncthreads();

    FragU af;
    {
      const v8u va = *(const v8u*)&As[colL * 32 + (lo ? 0 : 16)];
      const v8u vb = *(const v8u*)&As[colL * 32 + (lo ? 8 : 24)];
      af.u = __builtin_shufflevector(va, vb, 0, 1, 2, 3, 4, 5, 6, 7,
                                             8, 9, 10, 11, 12, 13, 14, 15);
    }
    const size_t chunkBase = whBase + ((size_t)((b * (NA_ / 32) + kc) * 8)) * 512;
#pragma unroll
    for (int nt = 0; nt < 8; ++nt) {
      FragU bf_;
      bf_.u = *(const v16u*)&Whp[chunkBase + (size_t)nt * 512 + laneoff];
      acc[nt] = __builtin_amdgcn_wmma_f32_16x16x32_bf16(
          false, af.b, false, bf_.b, (short)0, acc[nt], false, false);
    }
    __syncthreads();
  }

  // ELU + store (row-major bf16; feeds the next GEMM's A side)
#pragma unroll
  for (int nt = 0; nt < 8; ++nt) {
    int f = nt * 16 + colL;
#pragma unroll
    for (int r = 0; r < 8; ++r) {
      int n = n0 + r + (lo ? 0 : 8);
      float v = eluf(acc[nt][r]);
      out[(size_t)(b * NA_ + n) * outStride + h * outColOff + f] = f2bf(v);
    }
  }
}

// ---------------------------------------------------------------------------
// 11x11 direct conv (1 in / 1 out channel), pad 5, + bias + leaky.
// Block (64,4) computes a 4-row x 64-col tile via an LDS halo tile.
// ---------------------------------------------------------------------------
__global__ void k_conv(const float* __restrict__ X, float* __restrict__ Y,
                       const float* __restrict__ wgt, const float* __restrict__ bias,
                       int layer, int LA_, int W_) {
  __shared__ float tile[14 * 74];
  __shared__ float wk[121];
  const int tx = threadIdx.x, ty = threadIdx.y;
  const int tid = ty * 64 + tx;
  const int bc0 = blockIdx.x * 64, br0 = blockIdx.y * 4;
  const size_t base = (size_t)blockIdx.z * LA_ * W_;
  for (int i = tid; i < 14 * 74; i += 256) {
    int tr = i / 74, tc = i % 74;
    int gr = br0 + tr - 5, gc = bc0 + tc - 5;
    float v = 0.f;
    if (gr >= 0 && gr < LA_ && gc >= 0 && gc < W_) v = X[base + (size_t)gr * W_ + gc];
    tile[i] = v;
  }
  for (int i = tid; i < 121; i += 256) wk[i] = wgt[i];
  __syncthreads();
  float a = 0.f;
#pragma unroll
  for (int ky = 0; ky < 11; ++ky)
#pragma unroll
    for (int kx = 0; kx < 11; ++kx)
      a = fmaf(tile[(ty + ky) * 74 + tx + kx], wk[ky * 11 + kx], a);
  a = leakyf(a + bias[layer]);
  Y[base + (size_t)(br0 + ty) * W_ + (bc0 + tx)] = a;
}

// ---------------------------------------------------------------------------
// Masked mean over the sequence axis: out[b,c] = sum_l V[b,l,c]*m[b,l] / sum_l m
// ---------------------------------------------------------------------------
__global__ void k_masked_mean(const float* __restrict__ V, const float* __restrict__ mask,
                              float* __restrict__ out, int L) {
  const int b = blockIdx.x, c = threadIdx.x;
  float va = 0.f, ma = 0.f;
  for (int l = 0; l < L; ++l) {
    float m = mask[(size_t)b * L + l];
    va += V[((size_t)b * L + l) * 128 + c] * m;
    ma += m;
  }
  out[b * 128 + c] = va / ma;
}

// s2[b] = dot(vec[b,:], w) + bias
__global__ void k_vecdot(const float* __restrict__ vec, const float* __restrict__ w,
                         const float* __restrict__ bias, float* __restrict__ out) {
  __shared__ float red[128];
  int b = blockIdx.x, c = threadIdx.x;
  red[c] = vec[b * 128 + c] * w[c];
  __syncthreads();
  for (int s = 64; s; s >>= 1) { if (c < s) red[c] += red[c + s]; __syncthreads(); }
  if (c == 0) out[b] = red[0] + bias[0];
}

// V[row,:] *= tanh(dot(V[row,:], w1) + s2[b]); wave per row
__global__ void k_rowscale(float* __restrict__ V, const float* __restrict__ w1,
                           const float* __restrict__ s2, int L) {
  int wid = threadIdx.x >> 5, lane = threadIdx.x & 31;
  int row = blockIdx.x * (blockDim.x >> 5) + wid;
  int b = row / L;
  float s = 0.f;
#pragma unroll
  for (int j = 0; j < 4; ++j) s += V[(size_t)row * 128 + lane + 32 * j] * w1[lane + 32 * j];
#pragma unroll
  for (int off = 16; off; off >>= 1) s += __shfl_xor(s, off, 32);
  float sc = tanhf(s + s2[b]);
#pragma unroll
  for (int j = 0; j < 4; ++j) V[(size_t)row * 128 + lane + 32 * j] *= sc;
}

// out[b,k] = [comp|prot] . pred_w[:,k] + pred_b[k]
__global__ void k_predict(const float* __restrict__ comp, const float* __restrict__ prot,
                          const float* __restrict__ w, const float* __restrict__ bias,
                          float* __restrict__ out) {
  int t = threadIdx.x;
  if (t >= 64) return;
  int b = t >> 1, k = t & 1;
  float s = bias[k];
  for (int j = 0; j < 128; ++j) s += comp[b * 128 + j] * w[j * 2 + k];
  for (int j = 0; j < 128; ++j) s += prot[b * 128 + j] * w[(128 + j) * 2 + k];
  out[b * 2 + k] = s;
}

// ---------------------------------------------------------------------------
// Orchestration
// ---------------------------------------------------------------------------
extern "C" void kernel_launch(void* const* d_in, const int* in_sizes, int n_in,
                              void* d_out, int out_size, void* d_ws, size_t ws_size,
                              hipStream_t stream) {
  (void)in_sizes; (void)n_in; (void)out_size; (void)ws_size;
  constexpr int Bc = 32, NAc = 512, LAc = 4096, Cc = 128, Hc = 4;

  const int*   atoms      = (const int*)d_in[0];
  const float* atoms_mask = (const float*)d_in[1];
  const int*   adj        = (const int*)d_in[2];
  const int*   amino      = (const int*)d_in[3];
  const float* amino_mask = (const float*)d_in[4];
  const float* emb_atom   = (const float*)d_in[5];
  const float* emb_amino  = (const float*)d_in[6];
  const float* gat_W      = (const float*)d_in[7];
  const float* gat_a      = (const float*)d_in[8];
  const float* gat_out_W  = (const float*)d_in[9];
  const float* gat_out_a  = (const float*)d_in[10];
  const float* W_comp_w   = (const float*)d_in[11];
  const float* W_comp_b   = (const float*)d_in[12];
  const float* conv_w     = (const float*)d_in[13];
  const float* conv_b     = (const float*)d_in[14];
  const float* W_prot_w   = (const float*)d_in[15];
  const float* W_prot_b   = (const float*)d_in[16];
  const float* W_att_w    = (const float*)d_in[17];
  const float* W_att_b    = (const float*)d_in[18];
  const float* att_a2p_w  = (const float*)d_in[19];
  const float* att_a2p_b  = (const float*)d_in[20];
  const float* att_p2c_w  = (const float*)d_in[21];
  const float* att_p2c_b  = (const float*)d_in[22];
  const float* pred_w     = (const float*)d_in[23];
  const float* pred_b     = (const float*)d_in[24];
  float* outp = (float*)d_out;

  char* ws = (char*)d_ws;
  size_t off = 0;
  auto alloc = [&](size_t bytes) -> char* {
    char* p = ws + off;
    off = (off + bytes + 255) & ~(size_t)255;
    return p;
  };
  const size_t MA = (size_t)Bc * NAc;   // 16384 atom rows
  const size_t MP = (size_t)Bc * LAc;   // 131072 amino rows

  u16*   av_bf    = (u16*)alloc(MA * Cc * 2);
  u16*   gatWpk   = (u16*)alloc((size_t)Hc * Cc * Cc * 2);   // packed, 4 heads
  u16*   gatOWpk  = (u16*)alloc((size_t)4 * Cc * Cc * 2);    // packed 512x128
  u16*   Wcomppk  = (u16*)alloc((size_t)Cc * Cc * 2);
  u16*   Wprotpk  = (u16*)alloc((size_t)Cc * Cc * 2);
  u16*   Wattpk   = (u16*)alloc((size_t)Cc * Cc * 2);
  u16*   Wh1      = (u16*)alloc((size_t)Hc * MA * Cc * 2);   // row-major (for rowdot)
  u16*   Wh1p     = (u16*)alloc((size_t)Hc * MA * Cc * 2);   // packed (for attn WMMA)
  float* e1a      = (float*)alloc((size_t)Hc * MA * 4);
  float* e2a      = (float*)alloc((size_t)Hc * MA * 4);
  u16*   multi_bf = (u16*)alloc(MA * 4 * Cc * 2);
  u16*   Wh2      = (u16*)alloc(MA * Cc * 2);
  u16*   Wh2p     = (u16*)alloc(MA * Cc * 2);
  float* e1b      = (float*)alloc(MA * 4);
  float* e2b      = (float*)alloc(MA * 4);
  u16*   eluhp_bf = (u16*)alloc(MA * Cc * 2);
  u16*   av0_bf   = (u16*)alloc(MA * Cc * 2);
  float* atomsv   = (float*)alloc(MA * Cc * 4);
  float* x0       = (float*)alloc(MP * Cc * 4);
  float* x1       = (float*)alloc(MP * Cc * 4);
  float* protv    = (float*)alloc((size_t)Bc * Cc * 4);
  float* compv    = (float*)alloc((size_t)Bc * Cc * 4);
  float* protv2   = (float*)alloc((size_t)Bc * Cc * 4);
  float* s2a      = (float*)alloc(Bc * 4);
  float* s2p      = (float*)alloc(Bc * 4);
  // reuse x0 region after conv stack: first half bf16 conv-out, second half bf16 amino_v0
  u16* convbf  = (u16*)x0;
  u16* amv0_bf = (u16*)x0 + MP * Cc;
  float* aminov = x1;  // reuse x1 for fp32 amino_v

  // weight packing to WMMA B-fragment order (bf16)
  k_pack_w<<<dim3(64, Hc), 256, 0, stream>>>(gat_W, gatWpk, (long long)Cc * Cc,
                                             (long long)Cc * Cc, (long long)Cc * Cc);
  k_pack_w<<<dim3(256, 1), 256, 0, stream>>>(gat_out_W, gatOWpk, (long long)4 * Cc * Cc, 0LL, 0LL);
  k_pack_w<<<dim3(64, 1), 256, 0, stream>>>(W_comp_w, Wcomppk, (long long)Cc * Cc, 0LL, 0LL);
  k_pack_w<<<dim3(64, 1), 256, 0, stream>>>(W_prot_w, Wprotpk, (long long)Cc * Cc, 0LL, 0LL);
  k_pack_w<<<dim3(64, 1), 256, 0, stream>>>(W_att_w, Wattpk, (long long)Cc * Cc, 0LL, 0LL);

  // ---- compound GAT ----
  k_gather_bf<<<dim3((unsigned)MA), Cc, 0, stream>>>(atoms, emb_atom, av_bf, Cc);
  k_gemm_wmma<<<dim3((unsigned)(MA / 16), Hc), 32, 0, stream>>>(
      av_bf, gatWpk, nullptr, Wh1, nullptr, Wh1p, (int)MA, Cc,
      0LL, (long long)Cc * Cc, (long long)MA * Cc, 0);
  k_rowdot<<<dim3((unsigned)((Hc * MA) / 8)), 256, 0, stream>>>(
      Wh1, gat_a, e1a, e2a, (int)MA, 2 * Cc, (int)(Hc * MA));
  k_gat_attn<<<dim3(NAc / 16, Bc, Hc), 32, 0, stream>>>(
      Wh1p, e1a, e2a, adj, multi_bf, Hc * Cc, Cc, NAc);
  k_gemm_wmma<<<dim3((unsigned)(MA / 16), 1), 32, 0, stream>>>(
      multi_bf, gatOWpk, nullptr, Wh2, nullptr, Wh2p, (int)MA, Hc * Cc, 0LL, 0LL, 0LL, 0);
  k_rowdot<<<dim3((unsigned)(MA / 8)), 256, 0, stream>>>(
      Wh2, gat_out_a, e1b, e2b, 1 << 30, 0, (int)MA);
  k_gat_attn<<<dim3(NAc / 16, Bc, 1), 32, 0, stream>>>(
      Wh2p, e1b, e2b, adj, eluhp_bf, Cc, 0, NAc);
  k_gemm_wmma<<<dim3((unsigned)(MA / 16), 1), 32, 0, stream>>>(
      eluhp_bf, Wcomppk, W_comp_b, av0_bf, nullptr, nullptr, (int)MA, Cc, 0LL, 0LL, 0LL, 1);
  k_gemm_wmma<<<dim3((unsigned)(MA / 16), 1), 32, 0, stream>>>(
      av0_bf, Wattpk, W_att_b, nullptr, atomsv, nullptr, (int)MA, Cc, 0LL, 0LL, 0LL, 1);

  // ---- protein CNN ----
  k_gather_f32<<<dim3((unsigned)MP), Cc, 0, stream>>>(amino, emb_amino, x0, Cc);
  dim3 cgrid(Cc / 64, LAc / 4, Bc), cblk(64, 4);
  k_conv<<<cgrid, cblk, 0, stream>>>(x0, x1, conv_w,       conv_b, 0, LAc, Cc);
  k_conv<<<cgrid, cblk, 0, stream>>>(x1, x0, conv_w + 121, conv_b, 1, LAc, Cc);
  k_conv<<<cgrid, cblk, 0, stream>>>(x0, x1, conv_w + 242, conv_b, 2, LAc, Cc);
  k_f2bf<<<dim3(4096), 256, 0, stream>>>(x1, convbf, (long long)MP * Cc);
  k_gemm_wmma<<<dim3((unsigned)(MP / 16), 1), 32, 0, stream>>>(
      convbf, Wprotpk, W_prot_b, amv0_bf, nullptr, nullptr, (int)MP, Cc, 0LL, 0LL, 0LL, 1);
  k_gemm_wmma<<<dim3((unsigned)(MP / 16), 1), 32, 0, stream>>>(
      amv0_bf, Wattpk, W_att_b, nullptr, aminov, nullptr, (int)MP, Cc, 0LL, 0LL, 0LL, 1);

  // ---- bidirectional attention + prediction ----
  k_masked_mean<<<dim3(Bc), Cc, 0, stream>>>(aminov, amino_mask, protv, LAc);
  k_vecdot<<<dim3(Bc), Cc, 0, stream>>>(protv, att_a2p_w + Cc, att_a2p_b, s2a);
  k_rowscale<<<dim3((unsigned)(MA / 8)), 256, 0, stream>>>(atomsv, att_a2p_w, s2a, NAc);
  k_masked_mean<<<dim3(Bc), Cc, 0, stream>>>(atomsv, atoms_mask, compv, NAc);
  k_vecdot<<<dim3(Bc), Cc, 0, stream>>>(compv, att_p2c_w + Cc, att_p2c_b, s2p);
  k_rowscale<<<dim3((unsigned)(MP / 8)), 256, 0, stream>>>(aminov, att_p2c_w, s2p, LAc);
  k_masked_mean<<<dim3(Bc), Cc, 0, stream>>>(aminov, amino_mask, protv2, LAc);
  k_predict<<<1, 64, 0, stream>>>(compv, protv2, pred_w, pred_b, outp);
}
